// GCN_review_9268539425563
// MI455X (gfx1250) — matrix operand order repaired and checked by reading, not compiled, not run on hardware
//
#include <hip/hip_runtime.h>

// ---- CDNA5 types ----
typedef __attribute__((ext_vector_type(16))) __bf16 v16bf;
typedef __attribute__((ext_vector_type(8)))  float  v8f;
typedef __attribute__((ext_vector_type(4)))  float  v4f;
typedef __attribute__((ext_vector_type(4)))  int    v4i;

#define D_FEAT 64

// ---------------------------------------------------------------------------
// Kernel 0: zero the f32 accumulator in workspace (atomics need a clean slate)
// ---------------------------------------------------------------------------
__global__ void zero_acc_kernel(v4f* __restrict__ p, long n4) {
  long i = (long)blockIdx.x * blockDim.x + threadIdx.x;
  long stride = (long)gridDim.x * blockDim.x;
  v4f z = {0.f, 0.f, 0.f, 0.f};
  for (; i < n4; i += stride) p[i] = z;
}

// ---------------------------------------------------------------------------
// Kernel 1: fused   m = (rf * (ci[src]*drop)) @ W^T ;  acc[dst] += m
// One wave32 per 16-edge tile. bf16 WMMA with f32 accumulation.
//   A (16x32 bf16): rows = edges (pre-scaled), K = input feature d
//   B (32x16 bf16): B[k][n] = W[n][k]  (loop-invariant, built once per wave)
//   C (16x16 f32) x 4 feature tiles
// dst node IDs for this lane's 8 C rows are contiguous -> preloaded as two
// b128 vector loads issued BEFORE the WMMA chain so their latency hides.
// ---------------------------------------------------------------------------
__global__ void __launch_bounds__(256)
fused_edge_gemm_scatter(const float* __restrict__ rf,    // [E,64]
                        const float* __restrict__ ci,    // [N]
                        const float* __restrict__ W,     // [64,64] row-major
                        const float* __restrict__ drop,  // [E]
                        const int*   __restrict__ src,   // [E]
                        const int*   __restrict__ dst,   // [E]
                        float*       __restrict__ acc,   // [N,64] zeroed f32
                        int E)
{
  const int lane = threadIdx.x & 31;
  const int wave = threadIdx.x >> 5;
  const int wavesPerBlock = blockDim.x >> 5;
  const int m    = lane & 15;   // row-within-halfwave / column index
  const int half = lane >> 4;   // 0: lanes 0-15, 1: lanes 16-31

  // ---- Loop-invariant B fragments: lane n = ft*16 + m holds a contiguous
  //      16-float run of row (ft*16+m) of W (K = half*16.. / +32 for kstep 1).
  v16bf Bf[4][2];
#pragma unroll
  for (int ft = 0; ft < 4; ++ft) {
    const int wrow = ft * 16 + m;
#pragma unroll
    for (int ks = 0; ks < 2; ++ks) {
      const v4f* wp = reinterpret_cast<const v4f*>(W + wrow * D_FEAT + ks * 32 + half * 16);
      v16bf b;
#pragma unroll
      for (int q = 0; q < 4; ++q) {
        v4f w4 = wp[q];
#pragma unroll
        for (int i = 0; i < 4; ++i) b[q * 4 + i] = (__bf16)w4[i];
      }
      Bf[ft][ks] = b;
    }
  }

  const long nTiles     = ((long)E + 15) >> 4;
  const long tileStride = (long)gridDim.x * wavesPerBlock;
  long tile             = (long)blockIdx.x * wavesPerBlock + wave;

  for (; tile < nTiles; tile += tileStride) {
    const long eBase   = tile << 4;
    const bool full    = (eBase + 16 <= (long)E);
    const long rowBase = eBase + half * 8;         // first C row of this lane

    // ---- Early dst gather: 8 contiguous node IDs = two b128 loads.
    v4i nd0 = {0, 0, 0, 0}, nd1 = {0, 0, 0, 0};
    if (full) {
      const v4i* dp = reinterpret_cast<const v4i*>(dst + rowBase);
      nd0 = dp[0];
      nd1 = dp[1];
    }

    // ---- per-edge scalar: ci[src[e]] * drop_mask[e]
    const long e      = eBase + m;                 // this lane's A row (edge)
    const bool evalid = (e < (long)E);
    const long ec     = evalid ? e : (long)E - 1;
    const float se    = evalid ? (ci[src[ec]] * drop[ec]) : 0.0f;

    // prefetch next tile's streaming rows into GL2 (global_prefetch_b8)
    {
      const long en = (tile + tileStride) << 4;
      if (en < (long)E)
        __builtin_prefetch((const void*)(rf + (en + m) * D_FEAT), 0, 1);
    }

    // ---- A fragments, pre-scaled then converted f32 -> bf16 (NT loads).
    // A k order used here (same as B): lane<16 holds K = d0..d0+15
    //                  lane>=16 holds K = d0+16..d0+31
    const v4f* rp4 = reinterpret_cast<const v4f*>(rf + ec * D_FEAT);
    v16bf Af[2];
#pragma unroll
    for (int ks = 0; ks < 2; ++ks) {
      const int dbase = ks * 32 + half * 16;
      v4f c0 = __builtin_nontemporal_load(rp4 + (dbase >> 2));
      v4f c1 = __builtin_nontemporal_load(rp4 + (dbase >> 2) + 1);
      v4f c2 = __builtin_nontemporal_load(rp4 + ((dbase + 8) >> 2));
      v4f c3 = __builtin_nontemporal_load(rp4 + ((dbase + 8) >> 2) + 1);
      v16bf a;
#pragma unroll
      for (int i = 0; i < 4; ++i) {
        a[i]      = (__bf16)(c0[i] * se);
        a[4 + i]  = (__bf16)(c1[i] * se);
        a[8 + i]  = (__bf16)(c2[i] * se);
        a[12 + i] = (__bf16)(c3[i] * se);
      }
      Af[ks] = a;
    }

    // ---- 4 feature tiles x 2 K-steps = 8 v_wmma_f32_16x16x32_bf16
    v8f Cf[4];
#pragma unroll
    for (int ft = 0; ft < 4; ++ft) {
      v8f c = {};
      c = __builtin_amdgcn_wmma_f32_16x16x32_bf16(false, Af[0], false, Bf[ft][0],
                                                  (short)0, c, false, false);
      c = __builtin_amdgcn_wmma_f32_16x16x32_bf16(false, Af[1], false, Bf[ft][1],
                                                  (short)0, c, false, false);
      Cf[ft] = c;
    }

    // ---- Scatter-add straight from accumulator VGPRs.
    // C layout: lane holds col n = m (+16*ft); VGPR v holds row M = v + 8*half.
    if (full) {
      // fast path: no per-row guards, node IDs already in registers,
      // 32-bit offset math (N*64 fits u32); ft handled by inst offsets.
#pragma unroll
      for (int v = 0; v < 8; ++v) {
        const int node = (v < 4) ? nd0[v] : nd1[v - 4];
        float* outp = acc + ((unsigned)node * 64u + (unsigned)m);
#pragma unroll
        for (int ft = 0; ft < 4; ++ft)
          atomicAdd(outp + ft * 16, Cf[ft][v]);
      }
    } else {
      // tail path (at most one tile in the whole grid)
#pragma unroll
      for (int v = 0; v < 8; ++v) {
        const long er = rowBase + v;
        if (er < (long)E) {
          const int node = dst[er];
          float* outp = acc + ((unsigned)node * 64u + (unsigned)m);
#pragma unroll
          for (int ft = 0; ft < 4; ++ft)
            atomicAdd(outp + ft * 16, Cf[ft][v]);
        }
      }
    }
  }
}

// ---------------------------------------------------------------------------
// Kernel 2: out[n, :] = acc[n, :] * ci[n]
// ---------------------------------------------------------------------------
__global__ void scale_out_kernel(const v4f* __restrict__ acc,
                                 const float* __restrict__ ci,
                                 v4f* __restrict__ out, long n4) {
  long i = (long)blockIdx.x * blockDim.x + threadIdx.x;
  long stride = (long)gridDim.x * blockDim.x;
  for (; i < n4; i += stride) {
    const float c = ci[i >> 4];      // 16 x v4f per node row of 64 floats
    v4f v = acc[i];
    v4f r = {v[0] * c, v[1] * c, v[2] * c, v[3] * c};
    out[i] = r;
  }
}

// ---------------------------------------------------------------------------
extern "C" void kernel_launch(void* const* d_in, const int* in_sizes, int n_in,
                              void* d_out, int out_size, void* d_ws, size_t ws_size,
                              hipStream_t stream) {
  const float* rf   = (const float*)d_in[0];   // review_feat [E,64]
  const float* ci   = (const float*)d_in[1];   // [N]
  const float* W    = (const float*)d_in[2];   // [64,64]
  const float* drop = (const float*)d_in[3];   // [E]
  const int*   src  = (const int*)d_in[4];
  const int*   dst  = (const int*)d_in[5];

  const int E = in_sizes[0] / D_FEAT;
  const int N = in_sizes[1];

  float* acc = (float*)d_ws;                   // [N,64] f32 accumulator
  const long total  = (long)N * D_FEAT;
  const long total4 = total >> 2;

  // 1) zero accumulator
  zero_acc_kernel<<<2048, 256, 0, stream>>>((v4f*)acc, total4);

  // 2) fused GEMM + scatter  (grid-stride over 16-edge tiles, 8 waves/block)
  const long nTiles = ((long)E + 15) >> 4;
  int blocks = (int)((nTiles + 7) / 8);
  if (blocks > 2048) blocks = 2048;
  if (blocks < 1) blocks = 1;
  fused_edge_gemm_scatter<<<blocks, 256, 0, stream>>>(rf, ci, W, drop, src, dst, acc, E);

  // 3) final scale by ci into d_out
  scale_out_kernel<<<1024, 256, 0, stream>>>((const v4f*)acc, ci, (v4f*)d_out, total4);
}
